// Attention_56564719288642
// MI455X (gfx1250) — matrix operand order repaired and checked by reading, not compiled, not run on hardware
//
#include <hip/hip_runtime.h>
#include <hip/hip_bf16.h>

typedef __attribute__((ext_vector_type(16))) _Float16 v16h;
typedef __attribute__((ext_vector_type(8)))  _Float16 v8h;
typedef __attribute__((ext_vector_type(8)))  float    v8f;

#define HEADS 12
#define CDIM  768
#define DHEAD 64
#define TLEN  1025
#define TPAD  1088   // 68 q-tiles (divisible by 4 waves/block); key loop stops at 1056
#define NKB   33     // key blocks of 32 tokens

union U16H { v16h v; v8h h[2]; };

__device__ __forceinline__ v8f vzero8() {
  v8f z;
#pragma unroll
  for (int i = 0; i < 8; ++i) z[i] = 0.0f;
  return z;
}

// WMMA: D = A(16x32 f16) * B(32x16 f16) + C(16x16 f32)
__device__ __forceinline__ v8f wmma_f16(v16h a, v16h b, v8f c) {
  return __builtin_amdgcn_wmma_f32_16x16x32_f16(false, a, false, b, (short)0, c, false, false);
}

// A-fragment (16-bit A 16x32): lane holds row m=L%16;
// elems 0..15 = K kb..kb+15 contiguous (kb = (L/16)*16)
__device__ __forceinline__ v16h load_afrag(const _Float16* row, int kb) {
  U16H u;
  u.h[0] = *(const v8h*)(row + kb);
  u.h[1] = *(const v8h*)(row + kb + 8);
  return u.v;
}
// B-fragment: 16 contiguous f16 starting at K = (L/16)*16
__device__ __forceinline__ v16h load_bfrag(const _Float16* p) {
  U16H u;
  u.h[0] = *(const v8h*)p;
  u.h[1] = *(const v8h*)(p + 8);
  return u.v;
}

// butterfly max over the 16-lane groups sharing C/D-fragment rows
__device__ __forceinline__ float redmax16(float x) {
  x = fmaxf(x, __shfl_xor(x, 1));
  x = fmaxf(x, __shfl_xor(x, 2));
  x = fmaxf(x, __shfl_xor(x, 4));
  x = fmaxf(x, __shfl_xor(x, 8));
  return x;
}

// CDNA5 async global->LDS copy (ASYNCcnt path)
__device__ __forceinline__ void async_copy16(unsigned lds_off, const void* g) {
  asm volatile("global_load_async_to_lds_b128 %0, %1, off"
               :: "v"(lds_off), "v"((unsigned long long)(uintptr_t)g)
               : "memory");
}
__device__ __forceinline__ void wait_async0() {
  asm volatile("s_wait_asynccnt 0x0" ::: "memory");
}

// -------- Kernel 1: fused dwconv3x3 + BN for q,k,v; f16 in WMMA-friendly layouts
__global__ void __launch_bounds__(256) qkv_kernel(
    const float* __restrict__ x,
    const float* __restrict__ wq, const float* __restrict__ gq, const float* __restrict__ bq,
    const float* __restrict__ mq, const float* __restrict__ vq,
    const float* __restrict__ wk, const float* __restrict__ gk, const float* __restrict__ bk,
    const float* __restrict__ mk, const float* __restrict__ vk,
    const float* __restrict__ wv, const float* __restrict__ gv, const float* __restrict__ bv,
    const float* __restrict__ mv, const float* __restrict__ vv,
    _Float16* __restrict__ qo, _Float16* __restrict__ ko, _Float16* __restrict__ vo) {
  int idx = blockIdx.x * blockDim.x + threadIdx.x;
  if (idx >= 16 * TPAD * CDIM) return;
  int c = idx % CDIM;
  int t = (idx / CDIM) % TPAD;
  int b = idx / (CDIM * TPAD);

  float qf = 0.0f, kf = 0.0f, vf = 0.0f;
  if (t == 0) {
    float val = x[(size_t)b * TLEN * CDIM + c];   // cls token passthrough
    qf = kf = vf = val;
  } else if (t <= 1024) {
    int p = t - 1;
    int py = p >> 5, px = p & 31;
    float aq = 0.0f, ak = 0.0f, av = 0.0f;
#pragma unroll
    for (int dy = -1; dy <= 1; ++dy) {
      int yy = py + dy;
      if ((unsigned)yy >= 32u) continue;
#pragma unroll
      for (int dx = -1; dx <= 1; ++dx) {
        int xx = px + dx;
        if ((unsigned)xx >= 32u) continue;
        float xv = x[(size_t)b * TLEN * CDIM + (size_t)(1 + yy * 32 + xx) * CDIM + c];
        int wi = c * 9 + (dy + 1) * 3 + (dx + 1);
        aq += xv * wq[wi];
        ak += xv * wk[wi];
        av += xv * wv[wi];
      }
    }
    float sq = gq[c] * rsqrtf(vq[c] + 1e-5f); qf = aq * sq + (bq[c] - mq[c] * sq);
    float sk = gk[c] * rsqrtf(vk[c] + 1e-5f); kf = ak * sk + (bk[c] - mk[c] * sk);
    float sv = gv[c] * rsqrtf(vv[c] + 1e-5f); vf = av * sv + (bv[c] - mv[c] * sv);
  } // else padded token -> zeros

  const float scale = 0.03608439182435161f;  // 768^-0.5 folded into q
  int hh = c >> 6, dd = c & 63;
  size_t rowbase = ((size_t)(b * HEADS + hh) * TPAD + t) * DHEAD + dd;
  qo[rowbase] = (_Float16)(qf * scale);
  ko[rowbase] = (_Float16)kf;
  vo[((size_t)(b * HEADS + hh) * DHEAD + dd) * TPAD + t] = (_Float16)vf;  // [d][T]
}

// -------- Kernel 2: flash attention, double-buffered async K/V staging.
// Block = 4 waves on 4 consecutive q-tiles of one (b,h); one barrier per key-block.
// All 8 K/V B-fragments preloaded up front so ds latency hides behind softmax VALU.
__global__ void __launch_bounds__(128) attn_kernel(
    const _Float16* __restrict__ q, const _Float16* __restrict__ k,
    const _Float16* __restrict__ v, _Float16* __restrict__ o) {
  // elems: K0[0,2048) K1[2048,4096) V0[4096,6144) V1[6144,8192) prob[8192,10240)
  __shared__ __align__(32) _Float16 smem[10240];
  int tid   = threadIdx.x;
  int lane  = tid & 31;
  int wslot = tid >> 5;
  _Float16* pl = smem + 8192 + wslot * 512;      // per-wave 16x32 prob tile
  unsigned sbase = (unsigned)(uintptr_t)smem;    // low 32 bits = LDS byte offset

  int bh  = blockIdx.x / 17;                     // b*12 + h
  int qt  = (blockIdx.x % 17) * 4 + wslot;       // 68 q-tiles per (b,h)
  int half = lane >> 4;
  int n    = lane & 15;

  const _Float16* kmat = k + (size_t)bh * TPAD * DHEAD;
  const _Float16* vmat = v + (size_t)bh * DHEAD * TPAD;

  // Q A-fragments (scale pre-folded)
  const _Float16* qrow = q + ((size_t)bh * TPAD + qt * 16 + n) * DHEAD;
  v16h aq0 = load_afrag(qrow, half * 16);
  v16h aq1 = load_afrag(qrow, 32 + half * 16);

  v16h ones;  // all-ones B matrix: rowsum(P) via WMMA lands in C/D layout
#pragma unroll
  for (int e = 0; e < 16; ++e) ones[e] = (_Float16)1.0f;

  float rmax[8];
#pragma unroll
  for (int r = 0; r < 8; ++r) rmax[r] = -1e30f;
  v8f rsum = vzero8();
  v8f acc0 = vzero8(), acc1 = vzero8(), acc2 = vzero8(), acc3 = vzero8();

  // cooperative stage of one 32-token K tile (4KB) + V tile (4KB) into buffer sel
  auto stage = [&](int t0, int sel) {
    const char* ks = (const char*)(kmat + (size_t)t0 * DHEAD);
    unsigned kof = sbase + (unsigned)sel * 4096u;
    async_copy16(kof + tid * 16, ks + tid * 16);
    async_copy16(kof + (tid + 128) * 16, ks + (tid + 128) * 16);
    unsigned vof = sbase + 8192u + (unsigned)sel * 4096u;
#pragma unroll
    for (int c = 0; c < 2; ++c) {
      int chunk = tid + c * 128;
      int dr = chunk >> 2, part = chunk & 3;
      async_copy16(vof + chunk * 16,
                   (const void*)(vmat + (size_t)dr * TPAD + t0 + part * 8));
    }
  };

  stage(0, 0);
  for (int it = 0; it < NKB; ++it) {
    int t0  = it * 32;
    int sel = it & 1;
    wait_async0();        // this wave's tile `it` is in LDS
    __syncthreads();      // all waves done staging `it` AND done computing `it-1`
    if (it + 1 < NKB) stage(t0 + 32, sel ^ 1);  // overlaps compute below

    const _Float16* skt = smem + sel * 2048;          // [32 tok][64 d]
    const _Float16* svt = smem + 4096 + sel * 2048;   // [64 d][32 tok]

    // ---- preload ALL B-fragments for this tile (16 ds_load_b128, overlapped)
    const _Float16* kr0 = skt + n * DHEAD + half * 16;
    const _Float16* kr1 = skt + (16 + n) * DHEAD + half * 16;
    v16h bk00 = load_bfrag(kr0);
    v16h bk01 = load_bfrag(kr0 + 32);
    v16h bk10 = load_bfrag(kr1);
    v16h bk11 = load_bfrag(kr1 + 32);
    v16h bv0 = load_bfrag(svt + (0 * 16 + n) * 32 + half * 16);
    v16h bv1 = load_bfrag(svt + (1 * 16 + n) * 32 + half * 16);
    v16h bv2 = load_bfrag(svt + (2 * 16 + n) * 32 + half * 16);
    v16h bv3 = load_bfrag(svt + (3 * 16 + n) * 32 + half * 16);

    // ---- S = Q*K^T (two 16-col tiles)
    v8f s0 = vzero8(), s1 = vzero8();
    s0 = wmma_f16(aq0, bk00, s0);
    s0 = wmma_f16(aq1, bk01, s0);
    s1 = wmma_f16(aq0, bk10, s1);
    s1 = wmma_f16(aq1, bk11, s1);

    // mask padded key columns: only the final block holds any (uniform branch)
    if (t0 + 32 > TLEN) {
      if (t0 + n >= TLEN) {
#pragma unroll
        for (int r = 0; r < 8; ++r) s0[r] = -1e30f;
      }
      if (t0 + 16 + n >= TLEN) {
#pragma unroll
        for (int r = 0; r < 8; ++r) s1[r] = -1e30f;
      }
    }

    // ---- online softmax: block max (butterfly), rescale, stage probs in LDS
#pragma unroll
    for (int r = 0; r < 8; ++r) {
      float bm = redmax16(fmaxf(s0[r], s1[r]));
      float m2 = fmaxf(rmax[r], bm);
      float corr = __expf(rmax[r] - m2);
      rmax[r] = m2;
      float p0 = __expf(s0[r] - m2);
      float p1 = __expf(s1[r] - m2);
      rsum[r] *= corr;
      acc0[r] *= corr; acc1[r] *= corr; acc2[r] *= corr; acc3[r] *= corr;
      int row = half * 8 + r;
      pl[row * 32 + n]      = (_Float16)p0;
      pl[row * 32 + 16 + n] = (_Float16)p1;
    }

    // P as A-fragment (same-wave LDS in-order via DScnt)
    v16h pa = load_afrag(pl + n * 32, half * 16);

    // row sums via WMMA against all-ones B
    rsum = wmma_f16(pa, ones, rsum);

    // ---- O += P * V (V fragments already resident)
    acc0 = wmma_f16(pa, bv0, acc0);
    acc1 = wmma_f16(pa, bv1, acc1);
    acc2 = wmma_f16(pa, bv2, acc2);
    acc3 = wmma_f16(pa, bv3, acc3);
  }

  // ---- finalize: o[b][t][h*64+d] f16
  int b  = bh / HEADS;
  int hh = bh % HEADS;
#pragma unroll
  for (int r = 0; r < 8; ++r) {
    float inv = 1.0f / rsum[r];
    size_t row = (size_t)b * TPAD + qt * 16 + half * 8 + r;
    _Float16* op = o + row * CDIM + hh * DHEAD + n;
    op[0]  = (_Float16)(acc0[r] * inv);
    op[16] = (_Float16)(acc1[r] * inv);
    op[32] = (_Float16)(acc2[r] * inv);
    op[48] = (_Float16)(acc3[r] * inv);
  }
}

// -------- Kernel 3a: proj_w f32 -> f16 ([cout][cin] row-major == B-fragment layout)
__global__ void __launch_bounds__(256) convert_w(const float* __restrict__ w,
                                                 _Float16* __restrict__ wh) {
  int i = blockIdx.x * blockDim.x + threadIdx.x;
  if (i < CDIM * CDIM) wh[i] = (_Float16)w[i];
}

// -------- Kernel 3b: out = o @ proj_w.T + proj_b.
// Block (8 waves) computes a 64x128 tile: A(64x32) + B(128x32) K-chunks double-buffered
// in LDS via async copies; each wave owns 16 rows x 64 cols (4 WMMA accumulators).
__global__ void __launch_bounds__(256) proj_kernel(
    const _Float16* __restrict__ o, const _Float16* __restrict__ pw,
    const float* __restrict__ pb, float* __restrict__ out) {
  // elems: A0[0,2048) A1[2048,4096) B0[4096,8192) B1[8192,12288)
  __shared__ __align__(32) _Float16 smem[12288];
  int tid  = threadIdx.x;
  int lane = tid & 31;
  int w    = tid >> 5;
  unsigned sbase = (unsigned)(uintptr_t)smem;

  int ng = blockIdx.x % 6;               // N group: 128 cols
  int mg = blockIdx.x / 6;               // 0..271: 64-row group over [16*TPAD)
  size_t m0 = (size_t)mg * 64;
  int n0 = ng * 128;
  int rt = w & 3;                        // wave's row-tile within the 64 rows
  int cg = w >> 2;                       // wave's 64-col half of the 128 cols
  int half = lane >> 4, n = lane & 15;

  v8f acc0 = vzero8(), acc1 = vzero8(), acc2 = vzero8(), acc3 = vzero8();

  auto stage = [&](int kk, int sel) {
    {  // A chunk: 64 rows x 32 k -> 256 x 16B, 1 per thread
      int r = tid >> 2, part = tid & 3;
      async_copy16(sbase + (unsigned)sel * 4096u + tid * 16,
                   (const void*)(o + (m0 + r) * CDIM + kk + part * 8));
    }
#pragma unroll
    for (int c = 0; c < 2; ++c) {  // B chunk: 128 rows x 32 k -> 512 x 16B
      int chunk = tid + c * 256;
      int r = chunk >> 2, part = chunk & 3;
      async_copy16(sbase + 8192u + (unsigned)sel * 8192u + chunk * 16,
                   (const void*)(pw + (size_t)(n0 + r) * CDIM + kk + part * 8));
    }
  };

  stage(0, 0);
  for (int it = 0; it < 24; ++it) {
    int sel = it & 1;
    wait_async0();
    __syncthreads();
    if (it + 1 < 24) stage((it + 1) * 32, sel ^ 1);

    const _Float16* At = smem + sel * 2048;          // [64][32]
    const _Float16* Bt = smem + 4096 + sel * 4096;   // [128][32]
    // preload all fragments, then issue the 4 WMMAs
    v16h a   = load_afrag(At + (rt * 16 + n) * 32, half * 16);
    v16h bb0 = load_bfrag(Bt + (cg * 64 + 0 * 16 + n) * 32 + half * 16);
    v16h bb1 = load_bfrag(Bt + (cg * 64 + 1 * 16 + n) * 32 + half * 16);
    v16h bb2 = load_bfrag(Bt + (cg * 64 + 2 * 16 + n) * 32 + half * 16);
    v16h bb3 = load_bfrag(Bt + (cg * 64 + 3 * 16 + n) * 32 + half * 16);
    acc0 = wmma_f16(a, bb0, acc0);
    acc1 = wmma_f16(a, bb1, acc1);
    acc2 = wmma_f16(a, bb2, acc2);
    acc3 = wmma_f16(a, bb3, acc3);
  }

  int b = (int)(m0 / TPAD);              // 64 | TPAD -> whole block same batch
  int tbase = (int)(m0 % TPAD) + rt * 16 + half * 8;
#pragma unroll
  for (int j = 0; j < 4; ++j) {
    int col = n0 + cg * 64 + j * 16 + n;
    float bias = pb[col];
    const v8f* accp = (j == 0) ? &acc0 : (j == 1) ? &acc1 : (j == 2) ? &acc2 : &acc3;
#pragma unroll
    for (int r = 0; r < 8; ++r) {
      int t = tbase + r;
      if (t < TLEN)
        out[((size_t)b * TLEN + t) * CDIM + col] = (*accp)[r] + bias;
    }
  }
}

extern "C" void kernel_launch(void* const* d_in, const int* in_sizes, int n_in,
                              void* d_out, int out_size, void* d_ws, size_t ws_size,
                              hipStream_t stream) {
  (void)in_sizes; (void)n_in; (void)out_size; (void)ws_size;
  // setup_inputs() order: x, h, w, then (w,g,b,m,v) for q,k,v, then proj_w, proj_b
  const float* x  = (const float*)d_in[0];
  const float* wq = (const float*)d_in[3];
  const float* gq = (const float*)d_in[4];
  const float* bq = (const float*)d_in[5];
  const float* mq = (const float*)d_in[6];
  const float* vq = (const float*)d_in[7];
  const float* wk = (const float*)d_in[8];
  const float* gk = (const float*)d_in[9];
  const float* bk = (const float*)d_in[10];
  const float* mk = (const float*)d_in[11];
  const float* vk = (const float*)d_in[12];
  const float* wv = (const float*)d_in[13];
  const float* gv = (const float*)d_in[14];
  const float* bv = (const float*)d_in[15];
  const float* mv = (const float*)d_in[16];
  const float* vv = (const float*)d_in[17];
  const float* pw = (const float*)d_in[18];
  const float* pb = (const float*)d_in[19];

  char* ws = (char*)d_ws;
  const size_t QKV_BYTES = (size_t)16 * HEADS * TPAD * DHEAD * 2;  // 26,738,688
  _Float16* qws = (_Float16*)(ws);
  _Float16* kws = (_Float16*)(ws + QKV_BYTES);
  _Float16* vws = (_Float16*)(ws + 2 * QKV_BYTES);
  _Float16* ows = (_Float16*)(ws + 3 * QKV_BYTES);  // [16][TPAD][768] f16
  _Float16* pwh = (_Float16*)(ws + 4 * QKV_BYTES);  // proj_w f16

  qkv_kernel<<<(16 * TPAD * CDIM) / 256, 256, 0, stream>>>(
      x, wq, gq, bq, mq, vq, wk, gk, bk, mk, vk, wv, gv, bv, mv, vv, qws, kws, vws);
  convert_w<<<(CDIM * CDIM) / 256, 256, 0, stream>>>(pw, pwh);
  attn_kernel<<<(16 * HEADS) * 17, 128, 0, stream>>>(qws, kws, vws, ows);
  proj_kernel<<<(16 * (TPAD / 64)) * 6, 256, 0, stream>>>(ows, pwh, pb, (float*)d_out);
}
